// DyConv_17867063951540
// MI455X (gfx1250) — matrix operand order repaired and checked by reading, not compiled, not run on hardware
//
#include <hip/hip_runtime.h>

typedef __attribute__((ext_vector_type(16))) _Float16 v16h;
typedef __attribute__((ext_vector_type(8)))  _Float16 v8h;
typedef __attribute__((ext_vector_type(8)))  float    v8f;

#define C_CH 256

__device__ __forceinline__ float silu_f(float v) { return v / (1.0f + __expf(-v)); }

// ---------------------------------------------------------------------------
// f32 -> f16 conversion (pointwise weights)
// ---------------------------------------------------------------------------
__global__ void f32_to_f16_kernel(const float* __restrict__ in, _Float16* __restrict__ out, int n) {
    int i = blockIdx.x * blockDim.x + threadIdx.x;
    if (i < n) out[i] = (_Float16)in[i];
}

// ---------------------------------------------------------------------------
// Depthwise 3x3 conv + bias + SiLU, NCHW f32 input -> [n_spatial][C] f16 output
// (channel-contiguous layout == WMMA B-operand friendly)
// ---------------------------------------------------------------------------
__global__ void dwconv_silu_f16(const float* __restrict__ x,
                                const float* __restrict__ dww,  // [C][3][3] (pre-offset)
                                const float* __restrict__ dwb,  // [C]       (pre-offset)
                                _Float16* __restrict__ act,     // [8*Ho*Wo][C]
                                int Hi, int Wi, int Ho, int Wo, int stride, int total) {
    int i = blockIdx.x * blockDim.x + threadIdx.x;
    if (i >= total) return;
    int c  = i & (C_CH - 1);
    int n  = i >> 8;
    int HWo = Ho * Wo;
    int b  = n / HWo;
    int hw = n - b * HWo;
    int ho = hw / Wo;
    int wo = hw - ho * Wo;
    const float* w   = dww + c * 9;
    const float* xin = x + (long)(b * C_CH + c) * Hi * Wi;
    float s = dwb[c];
#pragma unroll
    for (int dy = 0; dy < 3; ++dy) {
        int hh = ho * stride - 1 + dy;
        if ((unsigned)hh >= (unsigned)Hi) continue;
#pragma unroll
        for (int dx = 0; dx < 3; ++dx) {
            int ww = wo * stride - 1 + dx;
            if ((unsigned)ww >= (unsigned)Wi) continue;
            s += w[dy * 3 + dx] * xin[hh * Wi + ww];
        }
    }
    act[(long)n * C_CH + c] = (_Float16)silu_f(s);
}

// ---------------------------------------------------------------------------
// Pointwise 1x1 conv as WMMA GEMM + bias + SiLU.
//   D[co=256][sp] = W[co][ci=256] x Act[ci][sp],  f16 in / f32 acc
// Block: 256 threads = 8 waves; wave w -> co [32w, 32w+32); 4 spatial tiles of 16.
// A frag (16x32 f16): lane l(0..15) row co0+l, K = k0..k0+15;
//                     lane l+16 same row, K = k0+16..k0+31.
// B frag (32x16 f16): lane n col sp0+n, K = k0..k0+15; lane n+16 K = k0+16..k0+31.
// D (16x16 f32): lane n col, VGPR r row r (lanes 0-15) / r+8 (lanes 16-31).
// ---------------------------------------------------------------------------
__global__ void pw_gemm_wmma_silu(const _Float16* __restrict__ act,  // [Nsp][256]
                                  const _Float16* __restrict__ wgt,  // [256][256] (co,ci)
                                  const float* __restrict__ bias,    // [256]
                                  float* __restrict__ out,           // NCHW f32
                                  int HW) {
    const int tid  = threadIdx.x;
    const int wave = tid >> 5;
    const int lane = tid & 31;
    const int ln   = lane & 15;
    const int hi   = lane >> 4;        // 0 or 1 (K-half select)
    const int co0  = wave * 32;
    const long sp0 = (long)blockIdx.x * 64;

    v8f acc[2][4];
#pragma unroll
    for (int t = 0; t < 2; ++t)
#pragma unroll
        for (int s = 0; s < 4; ++s)
#pragma unroll
            for (int r = 0; r < 8; ++r) acc[t][s][r] = 0.0f;

    const _Float16* a0 = wgt + (co0 + ln) * C_CH + hi * 16;
    const _Float16* a1 = wgt + (co0 + 16 + ln) * C_CH + hi * 16;
    const _Float16* bp = act + (sp0 + ln) * C_CH + hi * 16;

    for (int kk = 0; kk < 8; ++kk) {
        const int ci0 = kk * 32;
        // prefetch next K-slab of activations (global_prefetch_b8)
        if (kk < 7) __builtin_prefetch(bp + ci0 + 32, 0, 1);

        v16h A[2], B[4];
        {
            v8h lo = *(const v8h*)(a0 + ci0);
            v8h hh = *(const v8h*)(a0 + ci0 + 8);
#pragma unroll
            for (int i = 0; i < 8; ++i) { A[0][i] = lo[i]; A[0][i + 8] = hh[i]; }
            lo = *(const v8h*)(a1 + ci0);
            hh = *(const v8h*)(a1 + ci0 + 8);
#pragma unroll
            for (int i = 0; i < 8; ++i) { A[1][i] = lo[i]; A[1][i + 8] = hh[i]; }
        }
#pragma unroll
        for (int s = 0; s < 4; ++s)
            B[s] = *(const v16h*)(bp + (long)s * 16 * C_CH + ci0);

#pragma unroll
        for (int t = 0; t < 2; ++t)
#pragma unroll
            for (int s = 0; s < 4; ++s)
                acc[t][s] = __builtin_amdgcn_wmma_f32_16x16x32_f16(
                    false, A[t], false, B[s], (short)0, acc[t][s], false, false);
    }

    // epilogue: bias + SiLU, store NCHW (coalesced 64B runs along hw per lane group)
#pragma unroll
    for (int t = 0; t < 2; ++t) {
        const int cobase = co0 + 16 * t + hi * 8;
#pragma unroll
        for (int s = 0; s < 4; ++s) {
            long spi = sp0 + 16 * s + ln;
            int b  = (int)(spi / HW);
            int hw = (int)(spi - (long)b * HW);
            float* op = out + ((long)b * C_CH + cobase) * HW + hw;
#pragma unroll
            for (int r = 0; r < 8; ++r) {
                float v = acc[t][s][r] + bias[cobase + r];
                op[(long)r * HW] = silu_f(v);
            }
        }
    }
}

// ---------------------------------------------------------------------------
// Bilinear 2x upsample, align_corners=True, NCHW f32
// ---------------------------------------------------------------------------
__global__ void upsample2x_ac(const float* __restrict__ in, float* __restrict__ out,
                              int h, int w, int total) {
    int i = blockIdx.x * blockDim.x + threadIdx.x;
    if (i >= total) return;
    int W = 2 * w, H = 2 * h;
    int ox = i % W;
    int r  = i / W;
    int oy = r % H;
    int bc = r / H;
    float sy = (float)(h - 1) / (float)(H - 1);
    float sx = (float)(w - 1) / (float)(W - 1);
    float fy = oy * sy, fx = ox * sx;
    int y0 = (int)floorf(fy), x0 = (int)floorf(fx);
    int y1 = min(y0 + 1, h - 1), x1 = min(x0 + 1, w - 1);
    float wy = fy - (float)y0, wx = fx - (float)x0;
    const float* p = in + (long)bc * h * w;
    float v00 = p[y0 * w + x0], v01 = p[y0 * w + x1];
    float v10 = p[y1 * w + x0], v11 = p[y1 * w + x1];
    float r0 = v00 * (1.f - wx) + v01 * wx;
    float r1 = v10 * (1.f - wx) + v11 * wx;
    out[i] = r0 * (1.f - wy) + r1 * wy;
}

// ---------------------------------------------------------------------------
// Per-(b,c) spatial mean (deterministic tree reduction). grid = B*C
// ---------------------------------------------------------------------------
__global__ void chan_mean_kernel(const float* __restrict__ f, float* __restrict__ mean, int HW) {
    __shared__ float sm[256];
    int bc = blockIdx.x;
    int t  = threadIdx.x;
    const float* p = f + (long)bc * HW;
    float s = 0.f;
    for (int i = t; i < HW; i += 256) s += p[i];
    sm[t] = s;
    __syncthreads();
    for (int o = 128; o > 0; o >>= 1) {
        if (t < o) sm[t] += sm[t + o];
        __syncthreads();
    }
    if (t == 0) mean[bc] = sm[0] / (float)HW;
}

// ---------------------------------------------------------------------------
// Attention scalar per batch: hsig(relu(sum_c aw[c]*mean[b][c] + ab)). grid = 8
// ---------------------------------------------------------------------------
__global__ void attn_fin_kernel(const float* __restrict__ mean,  // [8][256]
                                const float* __restrict__ aw, const float* __restrict__ ab,
                                float* __restrict__ wout) {      // [8]
    __shared__ float sm[256];
    int b = blockIdx.x, t = threadIdx.x;
    sm[t] = mean[b * C_CH + t] * aw[t];
    __syncthreads();
    for (int o = 128; o > 0; o >>= 1) {
        if (t < o) sm[t] += sm[t + o];
        __syncthreads();
    }
    if (t == 0) {
        float a = fmaxf(sm[0] + ab[0], 0.f);
        wout[b] = fminf(a + 3.f, 6.f) * (1.f / 6.f);
    }
}

// ---------------------------------------------------------------------------
// Fuse: max over attention-weighted branches + 3x3 maxpool (-inf pad), -> d_out
// ---------------------------------------------------------------------------
__global__ void fuse_max_kernel(const float* __restrict__ f0, const float* __restrict__ f1,
                                const float* __restrict__ f2,
                                const float* __restrict__ w0, const float* __restrict__ w1,
                                const float* __restrict__ w2,
                                int nb, float* __restrict__ out, int H, int W, int total) {
    int i = blockIdx.x * blockDim.x + threadIdx.x;
    if (i >= total) return;
    int HW = H * W;
    int hw = i % HW;
    int bc = i / HW;
    int b  = bc >> 8;
    int h  = hw / W;
    int w  = hw - h * W;
    float a0 = w0[b];
    float a1 = (nb > 1) ? w1[b] : 0.f;
    float a2 = (nb > 2) ? w2[b] : 0.f;
    float best = -3.4e38f;
    long base = (long)bc * HW;
    for (int dy = -1; dy <= 1; ++dy) {
        int hh = h + dy;
        if ((unsigned)hh >= (unsigned)H) continue;
        for (int dx = -1; dx <= 1; ++dx) {
            int ww = w + dx;
            if ((unsigned)ww >= (unsigned)W) continue;
            long idx = base + hh * W + ww;
            float m = f0[idx] * a0;
            if (nb > 1) m = fmaxf(m, f1[idx] * a1);
            if (nb > 2) m = fmaxf(m, f2[idx] * a2);
            best = fmaxf(best, m);
        }
    }
    out[i] = best;
}

// ---------------------------------------------------------------------------
// DyReLU coefficients: mean[8][256] -> fc1(64) relu -> fc2(1024) hsig. grid = 8
// ---------------------------------------------------------------------------
__global__ void dyrelu_coef_kernel(const float* __restrict__ mean,
                                   const float* __restrict__ fc1w, const float* __restrict__ fc1b,
                                   const float* __restrict__ fc2w, const float* __restrict__ fc2b,
                                   float* __restrict__ coef) {    // [8][1024]
    __shared__ float y[256];
    __shared__ float z[64];
    int b = blockIdx.x, t = threadIdx.x;
    y[t] = mean[b * C_CH + t];
    __syncthreads();
    if (t < 64) {
        float s = fc1b[t];
        for (int c = 0; c < 256; ++c) s += fc1w[t * 256 + c] * y[c];
        z[t] = fmaxf(s, 0.f);
    }
    __syncthreads();
    for (int k = t; k < 1024; k += 256) {
        float s = fc2b[k];
        for (int j = 0; j < 64; ++j) s += fc2w[k * 64 + j] * z[j];
        coef[b * 1024 + k] = fminf(fmaxf(s + 3.f, 0.f), 6.f) * (1.f / 6.f);
    }
}

// ---------------------------------------------------------------------------
// Apply DyReLU in place on fused output
// ---------------------------------------------------------------------------
__global__ void dyrelu_apply_kernel(float* __restrict__ out, const float* __restrict__ coef,
                                    int HW, int total) {
    int i = blockIdx.x * blockDim.x + threadIdx.x;
    if (i >= total) return;
    int bc = i / HW;
    int b  = bc >> 8;
    int c  = bc & (C_CH - 1);
    const float* cf = coef + b * 1024;
    float x  = out[i];
    float a1 = (cf[c] - 0.5f) * 2.f + 1.f;
    float b1 = cf[256 + c] - 0.5f;
    float a2 = (cf[512 + c] - 0.5f) * 2.f;
    float b2 = cf[768 + c] - 0.5f;
    out[i] = fmaxf(x * a1 + b1, x * a2 + b2);
}

// ---------------------------------------------------------------------------
extern "C" void kernel_launch(void* const* d_in, const int* in_sizes, int n_in,
                              void* d_out, int out_size, void* d_ws, size_t ws_size,
                              hipStream_t stream) {
    const float* xin[3] = {(const float*)d_in[0], (const float*)d_in[1], (const float*)d_in[2]};
    const float* dww  = (const float*)d_in[3];
    const float* dwb  = (const float*)d_in[4];
    const float* pww  = (const float*)d_in[5];
    const float* pwb  = (const float*)d_in[6];
    const float* attw = (const float*)d_in[7];
    const float* attb = (const float*)d_in[8];
    const float* fc1w = (const float*)d_in[9];
    const float* fc1b = (const float*)d_in[10];
    const float* fc2w = (const float*)d_in[11];
    const float* fc2b = (const float*)d_in[12];

    char*  ws  = (char*)d_ws;
    size_t off = 0;
    auto alloc = [&](size_t bytes) -> char* {
        char* p = ws + off;
        off += (bytes + 255) & ~(size_t)255;
        return p;
    };

    _Float16* wf16 = (_Float16*)alloc((size_t)9 * 65536 * 2);

    // 7 DWConv instances: {widx, src, Hi, stride}
    const int widx[7] = {1, 1, 4, 5, 4, 7, 8};
    const int src[7]  = {0, 1, 1, 0, 2, 2, 1};
    const int Hi[7]   = {64, 32, 32, 64, 16, 16, 32};
    const int st[7]   = {1, 1, 1, 2, 1, 1, 2};
    int Ho[7], Nsp[7];
    _Float16* actb[7];
    float*    gout[7];
    for (int i = 0; i < 7; ++i) {
        Ho[i]  = Hi[i] / st[i];
        Nsp[i] = 8 * Ho[i] * Ho[i];
        actb[i] = (_Float16*)alloc((size_t)Nsp[i] * C_CH * 2);
        gout[i] = (float*)alloc((size_t)Nsp[i] * C_CH * 4);
    }
    float* up1    = (float*)alloc((size_t)8 * C_CH * 64 * 64 * 4);
    float* up4    = (float*)alloc((size_t)8 * C_CH * 32 * 32 * 4);
    float* meanBr = (float*)alloc((size_t)7 * 2048 * 4);
    float* meanFu = (float*)alloc((size_t)3 * 2048 * 4);
    float* attnw  = (float*)alloc((size_t)7 * 8 * 4);
    float* coef   = (float*)alloc((size_t)3 * 8 * 1024 * 4);

    // 1) pointwise weights -> f16
    {
        int n = 9 * 65536;
        f32_to_f16_kernel<<<(n + 255) / 256, 256, 0, stream>>>(pww, wf16, n);
    }
    // 2) depthwise 3x3 + SiLU -> f16 [n][ci]
    for (int i = 0; i < 7; ++i) {
        int total = Nsp[i] * C_CH;
        dwconv_silu_f16<<<(total + 255) / 256, 256, 0, stream>>>(
            xin[src[i]], dww + widx[i] * C_CH * 9, dwb + widx[i] * C_CH,
            actb[i], Hi[i], Hi[i], Ho[i], Ho[i], st[i], total);
    }
    // 3) pointwise 1x1 via WMMA + bias + SiLU -> f32 NCHW
    for (int i = 0; i < 7; ++i) {
        pw_gemm_wmma_silu<<<Nsp[i] / 64, 256, 0, stream>>>(
            actb[i], wf16 + (size_t)widx[i] * 65536, pwb + widx[i] * C_CH,
            gout[i], Ho[i] * Ho[i]);
    }
    // 4) bilinear upsamples (align_corners)
    {
        int t1 = 8 * C_CH * 64 * 64;
        upsample2x_ac<<<(t1 + 255) / 256, 256, 0, stream>>>(gout[1], up1, 32, 32, t1);
        int t4 = 8 * C_CH * 32 * 32;
        upsample2x_ac<<<(t4 + 255) / 256, 256, 0, stream>>>(gout[4], up4, 16, 16, t4);
    }
    // Final branch tensors per level
    const float* fb[7]  = {gout[0], up1, gout[2], gout[3], up4, gout[5], gout[6]};
    const int    fbHW[7] = {4096, 4096, 1024, 1024, 1024, 256, 256};
    // 5) per-(b,c) means of branches, 6) attention scalars
    for (int i = 0; i < 7; ++i)
        chan_mean_kernel<<<2048, 256, 0, stream>>>(fb[i], meanBr + i * 2048, fbHW[i]);
    for (int i = 0; i < 7; ++i)
        attn_fin_kernel<<<8, 256, 0, stream>>>(meanBr + i * 2048, attw, attb, attnw + i * 8);
    // 7) fuse (branch max + 3x3 maxpool) into d_out
    float* o0 = (float*)d_out;
    float* o1 = o0 + (size_t)8 * C_CH * 4096;
    float* o2 = o1 + (size_t)8 * C_CH * 1024;
    int tot0 = 8 * C_CH * 4096, tot1 = 8 * C_CH * 1024, tot2 = 8 * C_CH * 256;
    fuse_max_kernel<<<(tot0 + 255) / 256, 256, 0, stream>>>(
        fb[0], fb[1], fb[1], attnw + 0, attnw + 8, attnw + 8, 2, o0, 64, 64, tot0);
    fuse_max_kernel<<<(tot1 + 255) / 256, 256, 0, stream>>>(
        fb[2], fb[3], fb[4], attnw + 16, attnw + 24, attnw + 32, 3, o1, 32, 32, tot1);
    fuse_max_kernel<<<(tot2 + 255) / 256, 256, 0, stream>>>(
        fb[5], fb[6], fb[6], attnw + 40, attnw + 48, attnw + 48, 2, o2, 16, 16, tot2);
    // 8) fused means, 9) DyReLU coefficients, 10) apply in place
    float* olv[3] = {o0, o1, o2};
    int    oHW[3] = {4096, 1024, 256};
    int    otot[3] = {tot0, tot1, tot2};
    for (int l = 0; l < 3; ++l)
        chan_mean_kernel<<<2048, 256, 0, stream>>>(olv[l], meanFu + l * 2048, oHW[l]);
    for (int l = 0; l < 3; ++l)
        dyrelu_coef_kernel<<<8, 256, 0, stream>>>(meanFu + l * 2048, fc1w, fc1b, fc2w, fc2b,
                                                  coef + l * 8192);
    for (int l = 0; l < 3; ++l)
        dyrelu_apply_kernel<<<(otot[l] + 255) / 256, 256, 0, stream>>>(olv[l], coef + l * 8192,
                                                                       oHW[l], otot[l]);
}
